// CascadedGroupAttention_15058155340301
// MI455X (gfx1250) — matrix operand order repaired and checked by reading, not compiled, not run on hardware
//
#include <hip/hip_runtime.h>
#include <hip/hip_bf16.h>

typedef __attribute__((ext_vector_type(16))) _Float16 v16h;
typedef __attribute__((ext_vector_type(8)))  _Float16 v8h;
typedef __attribute__((ext_vector_type(8)))  float    v8f;

#define CH_  384
#define NH_  8
#define ATT_ 48
#define QK_  16
#define VD_  48
#define RES_ 5
#define N_   25
#define NPAD 32

// Workspace layout:
//   pw_h : [384][384] f16  (converted projection weight)
//   y_h  : [B][NPAD=32][384] f16 (relu'd concat, channel-contiguous, n-padded w/ zeros)
static const size_t PW_BYTES = (size_t)CH_ * CH_ * sizeof(_Float16);
static const size_t YH_OFF   = ((PW_BYTES + 255) / 256) * 256;

// ---------------------------------------------------------------------------
// Kernel 0: f32 -> f16 convert of the (shared) projection weight.
// ---------------------------------------------------------------------------
__global__ void cvt_pw(const float* __restrict__ pw, _Float16* __restrict__ pwh) {
    int i = blockIdx.x * blockDim.x + threadIdx.x;
    if (i < CH_ * CH_) pwh[i] = (_Float16)pw[i];
}

// ---------------------------------------------------------------------------
// Kernel 1: cascaded group attention, one block per batch element.
// All intermediates live in LDS (~37 KB of the 320 KB WGP pool).
// Emits relu(concat(outs)) as f16 into y_h in WMMA-B-fragment-friendly layout.
// ---------------------------------------------------------------------------
__global__ __launch_bounds__(128)
void cascade(const float* __restrict__ x,
             const float* __restrict__ qw,  const float* __restrict__ qb,
             const float* __restrict__ dww, const float* __restrict__ dwb,
             const float* __restrict__ kw,  const float* __restrict__ kb,
             const float* __restrict__ vw,  const float* __restrict__ vb,
             const float* __restrict__ ab,  const int* __restrict__ rp,
             _Float16* __restrict__ yh) {
    __shared__ float att[ATT_][26];   // current cascaded activation (48 x 25)
    __shared__ float qpre[QK_][26];   // q before depthwise conv
    __shared__ float qv[QK_][26];     // q after depthwise conv
    __shared__ float kv[QK_][26];
    __shared__ float vv[VD_][26];
    __shared__ float pmat[N_][26];    // logits -> softmax probs
    __shared__ float bmat[N_][26];    // rel-pos bias expanded per head
    __shared__ float wq[QK_ * ATT_];
    __shared__ float wk[QK_ * ATT_];
    __shared__ float wv[VD_ * ATT_];
    __shared__ float wd[QK_ * 25];

    const int b   = blockIdx.x;
    const int tid = threadIdx.x;
    const int T   = 128;
    _Float16* yb = yh + (size_t)b * NPAD * CH_;

    // Zero the n = 25..31 padding rows (read by WMMA B-fragments, contribute 0).
    for (int idx = tid; idx < (NPAD - N_) * CH_; idx += T) {
        int n = N_ + idx / CH_;
        int c = idx % CH_;
        yb[(size_t)n * CH_ + c] = (_Float16)0.f;
    }

    for (int head = 0; head < NH_; ++head) {
        // --- stage 0: stage per-head weights, expand rel-pos bias, update att
        for (int idx = tid; idx < QK_ * ATT_; idx += T) {
            wq[idx] = qw[head * QK_ * ATT_ + idx];
            wk[idx] = kw[head * QK_ * ATT_ + idx];
        }
        for (int idx = tid; idx < VD_ * ATT_; idx += T)
            wv[idx] = vw[head * VD_ * ATT_ + idx];
        for (int idx = tid; idx < QK_ * 25; idx += T)
            wd[idx] = dww[head * QK_ * 25 + idx];
        for (int idx = tid; idx < N_ * N_; idx += T) {
            int n = idx / N_, m = idx % N_;
            bmat[n][m] = ab[head * N_ + rp[idx]];
        }
        for (int idx = tid; idx < ATT_ * N_; idx += T) {
            int c = idx / N_, n = idx % N_;
            float xv = x[((size_t)b * CH_ + head * ATT_ + c) * N_ + n];
            att[c][n] = (head == 0) ? xv : (att[c][n] + xv);
        }
        __syncthreads();

        // --- stage A: q-pre / k / v 1x1 convs (K = 48), fused work list
        for (int idx = tid; idx < 2 * QK_ * N_ + VD_ * N_; idx += T) {
            if (idx < QK_ * N_) {
                int d = idx / N_, n = idx % N_;
                float s = qb[head * QK_ + d];
                for (int c = 0; c < ATT_; ++c) s += wq[d * ATT_ + c] * att[c][n];
                qpre[d][n] = s;
            } else if (idx < 2 * QK_ * N_) {
                int j = idx - QK_ * N_;
                int d = j / N_, n = j % N_;
                float s = kb[head * QK_ + d];
                for (int c = 0; c < ATT_; ++c) s += wk[d * ATT_ + c] * att[c][n];
                kv[d][n] = s;
            } else {
                int j = idx - 2 * QK_ * N_;
                int e = j / N_, n = j % N_;
                float s = vb[head * VD_ + e];
                for (int c = 0; c < ATT_; ++c) s += wv[e * ATT_ + c] * att[c][n];
                vv[e][n] = s;
            }
        }
        __syncthreads();

        // --- stage B: 5x5 depthwise conv (SAME) on q
        for (int idx = tid; idx < QK_ * N_; idx += T) {
            int d = idx / N_, n = idx % N_;
            int h = n / RES_, w = n % RES_;
            float s = dwb[head * QK_ + d];
            for (int kh = 0; kh < 5; ++kh) {
                int hh = h + kh - 2;
                if (hh < 0 || hh >= RES_) continue;
                for (int kw2 = 0; kw2 < 5; ++kw2) {
                    int ww = w + kw2 - 2;
                    if (ww < 0 || ww >= RES_) continue;
                    s += qpre[d][hh * RES_ + ww] * wd[d * 25 + kh * 5 + kw2];
                }
            }
            qv[d][n] = s;
        }
        __syncthreads();

        // --- stage C: logits = q^T k * 1/sqrt(16) + bias
        for (int idx = tid; idx < N_ * N_; idx += T) {
            int n = idx / N_, m = idx % N_;
            float s = 0.f;
            for (int d = 0; d < QK_; ++d) s += qv[d][n] * kv[d][m];
            pmat[n][m] = s * 0.25f + bmat[n][m];
        }
        __syncthreads();

        // --- stage D: row softmax (one 25-wide row per thread)
        if (tid < N_) {
            float mx = pmat[tid][0];
            for (int m = 1; m < N_; ++m) mx = fmaxf(mx, pmat[tid][m]);
            float sum = 0.f;
            for (int m = 0; m < N_; ++m) {
                float e = __expf(pmat[tid][m] - mx);
                pmat[tid][m] = e;
                sum += e;
            }
            float inv = 1.f / sum;
            for (int m = 0; m < N_; ++m) pmat[tid][m] *= inv;
        }
        __syncthreads();

        // --- stage E: att_out = v @ p^T ; write relu(att_out) to y_h (f16)
        for (int idx = tid; idx < VD_ * N_; idx += T) {
            int e = idx / N_, n = idx % N_;
            float s = 0.f;
            for (int m = 0; m < N_; ++m) s += vv[e][m] * pmat[n][m];
            att[e][n] = s;                                  // cascade input for next head
            yb[(size_t)n * CH_ + head * VD_ + e] = (_Float16)fmaxf(s, 0.f);
        }
        __syncthreads();
    }
}

// ---------------------------------------------------------------------------
// Kernel 2: out[b] = p_w (384x384) @ y[b] (384x25) + p_b, via WMMA f16->f32.
// One block (8 waves) per batch; each wave: 3 M-tiles x 2 N-tiles, K = 12x32.
// Fragment layouts used here (wave32, contiguous k in A and B):
//   A (16x32 f16): lane L holds M=L%16; halves 0..7 -> K=16*(L/16)+h,
//                  halves 8..15 -> K=16*(L/16)+h   (two 16B loads)
//   B (32x16 f16): lane L holds N=L%16; half h -> K=16*(L/16)+h (one 32B load)
//   D (16x16 f32): VGPR r -> M=r+8*(L/16), N=L%16
// ---------------------------------------------------------------------------
__global__ __launch_bounds__(256)
void proj_wmma(const _Float16* __restrict__ pwh, const _Float16* __restrict__ yh,
               const float* __restrict__ pb, float* __restrict__ out) {
    const int b    = blockIdx.x;
    const int lane = threadIdx.x & 31;
    const int wave = threadIdx.x >> 5;
    const int hi   = lane >> 4;   // half-wave select
    const int ln   = lane & 15;

    const v8f vzero = {};
    v8f acc[3][2];
#pragma unroll
    for (int t = 0; t < 3; ++t)
#pragma unroll
        for (int nt = 0; nt < 2; ++nt) acc[t][nt] = vzero;

    const _Float16* yb = yh + (size_t)b * NPAD * CH_;

    for (int k0 = 0; k0 < CH_; k0 += 32) {
        // B fragments: channels contiguous in y_h -> single 32B load per lane
        v16h bfrag[2];
#pragma unroll
        for (int nt = 0; nt < 2; ++nt) {
            const _Float16* p = yb + (size_t)(nt * 16 + ln) * CH_ + k0 + 16 * hi;
            bfrag[nt] = *(const v16h*)p;
        }
#pragma unroll
        for (int t = 0; t < 3; ++t) {
            const int ot = wave * 3 + t;
            const int o  = ot * 16 + ln;
            const _Float16* p0 = pwh + (size_t)o * CH_ + k0 + 16 * hi;
            union { v16h v; v8h h[2]; } af;
            af.h[0] = *(const v8h*)p0;          // K = 16*hi + 0..7
            af.h[1] = *(const v8h*)(p0 + 8);    // K = 16*hi + 8..15
#pragma unroll
            for (int nt = 0; nt < 2; ++nt) {
                acc[t][nt] = __builtin_amdgcn_wmma_f32_16x16x32_f16(
                    /*neg_a=*/false, af.v, /*neg_b=*/false, bfrag[nt],
                    /*c_mod=*/(short)0, acc[t][nt],
                    /*reuse_a=*/false, /*reuse_b=*/false);
            }
        }
        if (k0 + 32 < CH_) {
            // global_prefetch_b8 for the next K-slab of B
            __builtin_prefetch((const void*)(yb + (size_t)ln * CH_ + k0 + 32), 0, 0);
        }
    }

    // Epilogue: scatter D fragments + bias, skip N padding (n >= 25)
#pragma unroll
    for (int t = 0; t < 3; ++t) {
        const int ot = wave * 3 + t;
#pragma unroll
        for (int nt = 0; nt < 2; ++nt) {
            int n = nt * 16 + ln;
            if (n < N_) {
#pragma unroll
                for (int r = 0; r < 8; ++r) {
                    int o = ot * 16 + r + 8 * hi;
                    out[((size_t)b * CH_ + o) * N_ + n] = acc[t][nt][r] + pb[o];
                }
            }
        }
    }
}

// ---------------------------------------------------------------------------
extern "C" void kernel_launch(void* const* d_in, const int* in_sizes, int n_in,
                              void* d_out, int out_size, void* d_ws, size_t ws_size,
                              hipStream_t stream) {
    (void)n_in; (void)out_size;
    const float* x   = (const float*)d_in[0];
    const float* qw  = (const float*)d_in[1];
    const float* qb  = (const float*)d_in[2];
    const float* dww = (const float*)d_in[3];
    const float* dwb = (const float*)d_in[4];
    const float* kw  = (const float*)d_in[5];
    const float* kb  = (const float*)d_in[6];
    const float* vw  = (const float*)d_in[7];
    const float* vb  = (const float*)d_in[8];
    const float* pw  = (const float*)d_in[9];
    const float* pb  = (const float*)d_in[10];
    const float* ab  = (const float*)d_in[11];
    const int*   rp  = (const int*)d_in[12];
    float* out = (float*)d_out;

    const int B = in_sizes[0] / (CH_ * N_);   // 128

    _Float16* pwh = (_Float16*)d_ws;
    _Float16* yh  = (_Float16*)((char*)d_ws + YH_OFF);
    // ws usage: 295KB (pw_h) + B*32*384*2B (~3.1MB for B=128)
    (void)ws_size;

    cvt_pw<<<(CH_ * CH_ + 255) / 256, 256, 0, stream>>>(pw, pwh);
    cascade<<<B, 128, 0, stream>>>(x, qw, qb, dww, dwb, kw, kb, vw, vb, ab, rp, yh);
    proj_wmma<<<B, 256, 0, stream>>>(pwh, yh, pb, out);
}
